// SelfSupervisedMamba_33921651704436
// MI455X (gfx1250) — matrix operand order repaired and checked by reading, not compiled, not run on hardware
//
#include <hip/hip_runtime.h>
#include <hip/hip_bf16.h>

// ---------------- constants for this problem ----------------
#define BB 8
#define LL 1024
#define DMODEL 256
#define DINNER 512
#define DSTATE 64
#define DTRANK 16
#define NXZ    1024                 // 2*DINNER
#define NXD    144                  // DTRANK + 2*DSTATE
#define MROWS  (BB*LL)              // 8192
#define SCAN_TPB 128                // channels per scan block

typedef __attribute__((ext_vector_type(16))) _Float16 v16h;
typedef __attribute__((ext_vector_type(8)))  _Float16 v8h;
typedef __attribute__((ext_vector_type(8)))  float    v8f;

// ---------------- f32 -> f16 convert ----------------
__global__ void k_f32_to_f16(const float* __restrict__ in, _Float16* __restrict__ out, int n) {
    int i = blockIdx.x * 256 + threadIdx.x;
    if (i < n) out[i] = (_Float16)in[i];
}

// pad dt_proj_w (512x16 f32) -> (512x32 f16, upper K zero)
__global__ void k_pad_dtw(const float* __restrict__ w, _Float16* __restrict__ w16) {
    int i = blockIdx.x * 256 + threadIdx.x;          // over 512*32
    if (i >= DINNER * 32) return;
    int k = i & 31, n = i >> 5;
    w16[i] = (k < DTRANK) ? (_Float16)w[n * DTRANK + k] : (_Float16)0.0f;
}

// extract dt columns from x_dbl (Mx144 f32) -> (Mx32 f16, upper K zero)
__global__ void k_extract_dt(const float* __restrict__ xdbl, _Float16* __restrict__ dt16) {
    int i = blockIdx.x * 256 + threadIdx.x;          // over M*32
    if (i >= MROWS * 32) return;
    int k = i & 31, m = i >> 5;
    dt16[i] = (k < DTRANK) ? (_Float16)xdbl[(size_t)m * NXD + k] : (_Float16)0.0f;
}

// ---------------- WMMA GEMM: out[M,N] = A[M,K] (f16) * W[N,K]^T (f16) ----------------
// One wave computes a 16 x (NT*16) output strip.  Per K-step: phase 1 issues the A
// fragment load + all NT B fragment loads as one clause (many loads in flight),
// phase 2 issues NT back-to-back v_wmma ops -> scheduler can use partial
// s_wait_loadcnt values and overlap loads with MMA.
// act: 0 = none, 1 = softplus.  Writes f32 (Cf) or f16 (Ch).
template <int NT>
__global__ __launch_bounds__(256) void k_wmma_gemm_bt(
    const _Float16* __restrict__ A, const _Float16* __restrict__ W,
    float* __restrict__ Cf, _Float16* __restrict__ Ch,
    const float* __restrict__ bias, int M, int N, int K, int act)
{
    const int lane    = threadIdx.x & 31;
    const int wave    = threadIdx.x >> 5;
    const int ntn     = N >> 4;                      // 16-wide tiles along N (multiple of NT)
    const int nstrips = ntn / NT;
    const int tiles   = (M >> 4) * nstrips;
    const int tile    = blockIdx.x * 8 + wave;       // uniform per wave
    if (tile >= tiles) return;                       // whole-wave exit, EXEC stays full

    const int tm   = (tile / nstrips) << 4;
    const int tn0  = (tile % nstrips) * (NT << 4);
    const int hi   = lane >> 4;                      // 0 | 1
    const int l15  = lane & 15;
    const int arow = tm + l15;                       // A row this lane feeds
    const int aklo = hi << 4;                        // A frag: K offset 0 or 16
    const int bklo = hi << 4;                        // B frag: K offset 0 or 16

    v8f acc[NT];
#pragma unroll
    for (int j = 0; j < NT; ++j) acc[j] = (v8f){};

    const _Float16* aprow = A + (size_t)arow * K;
    for (int kb = 0; kb < K; kb += 32) {
        // ---- phase 1: issue all fragment loads for this K step ----
        const _Float16* ap = aprow + kb + aklo;
        v8h a0 = *(const v8h*)(ap);                  // K = kb+aklo+0..7
        v8h a1 = *(const v8h*)(ap + 8);              // K = kb+aklo+8..15
        v16h bf[NT];
#pragma unroll
        for (int j = 0; j < NT; ++j) {
            const int wrow = tn0 + (j << 4) + l15;   // W row = output column
            bf[j] = *(const v16h*)(W + (size_t)wrow * K + kb + bklo);
        }
        if (kb + 32 < K) {                           // warm WGP$ for next K block
            __builtin_prefetch((const void*)(ap + 32), 0, 3);
        }
        v16h af;
#pragma unroll
        for (int i = 0; i < 8; ++i) { af[i] = a0[i]; af[i + 8] = a1[i]; }
        // ---- phase 2: NT back-to-back WMMAs reusing the A fragment ----
#pragma unroll
        for (int j = 0; j < NT; ++j) {
            acc[j] = __builtin_amdgcn_wmma_f32_16x16x32_f16(
                /*neg_a=*/false, af, /*neg_b=*/false, bf[j],
                /*c_mod=*/(short)0, acc[j], /*reuse_a=*/false, /*reuse_b=*/false);
        }
    }

#pragma unroll
    for (int j = 0; j < NT; ++j) {
        const int ncol = tn0 + (j << 4) + l15;
        float bv = bias ? bias[ncol] : 0.0f;
#pragma unroll
        for (int r = 0; r < 8; ++r) {
            int m = tm + r + (hi << 3);              // C/D layout: vgpr r -> M=r (+8 for hi lanes)
            float v = acc[j][r] + bv;
            if (act == 1) v = (v > 20.0f) ? v : __logf(1.0f + __expf(v));   // softplus
            size_t idx = (size_t)m * N + ncol;
            if (Cf) Cf[idx] = v;
            else    Ch[idx] = (_Float16)v;
        }
    }
}

// ---------------- causal depthwise conv1d(k=4) + bias + SiLU ----------------
__global__ void k_conv_silu(const float* __restrict__ xz, const float* __restrict__ cw,
                            const float* __restrict__ cb,
                            float* __restrict__ u32, _Float16* __restrict__ u16)
{
    int i = blockIdx.x * 256 + threadIdx.x;          // over B*L*DINNER
    if (i >= MROWS * DINNER) return;
    int d = i & (DINNER - 1);
    int t = (i >> 9) & (LL - 1);
    int b = i >> 19;
    float acc = cb[d];
#pragma unroll
    for (int j = 0; j < 4; ++j) {
        int tt = t - 3 + j;
        if (tt >= 0) acc += xz[((size_t)b * LL + tt) * NXZ + d] * cw[d * 4 + j];
    }
    float s = acc * (1.0f / (1.0f + __expf(-acc)));  // SiLU
    u32[i] = s;
    u16[i] = (_Float16)s;
}

// ---------------- sequential SSM scan, fused skip + gate ----------------
// grid = B * (DINNER/SCAN_TPB) blocks; thread = one d_inner channel.
// h[64] state + A row live in VGPRs for the whole 1024-step scan; B_t/C_t via LDS.
__global__ __launch_bounds__(SCAN_TPB, 1) void k_ssm_scan(
    const float* __restrict__ delta, const float* __restrict__ uconv,
    const float* __restrict__ xdbl,  const float* __restrict__ xz,
    const float* __restrict__ A_log, const float* __restrict__ Dp,
    _Float16* __restrict__ y16)
{
    const int blocks_per_b = DINNER / SCAN_TPB;      // 4
    const int b   = blockIdx.x / blocks_per_b;
    const int d   = (blockIdx.x % blocks_per_b) * SCAN_TPB + threadIdx.x;
    const int tid = threadIdx.x;                     // 0..127 == 2*DSTATE lanes for staging
    __shared__ float sB[DSTATE];
    __shared__ float sC[DSTATE];

    float h[DSTATE];
    float a[DSTATE];
#pragma unroll
    for (int s = 0; s < DSTATE; ++s) {
        h[s] = 0.0f;
        a[s] = -__expf(A_log[d * DSTATE + s]);       // A = -exp(A_log)
    }
    const float Dd = Dp[d];

    for (int t = 0; t < LL; ++t) {
        size_t row = (size_t)b * LL + t;
        {
            // B at cols [16,80), C at cols [80,144): both are xdbl[row*144 + 16 + tid]
            float v = xdbl[row * NXD + DTRANK + tid];
            if (tid < DSTATE) sB[tid] = v; else sC[tid - DSTATE] = v;
        }
        __syncthreads();

        float dt = delta[row * DINNER + d];
        float u  = uconv[row * DINNER + d];
        float z  = xz[row * NXZ + DINNER + d];
        float du = dt * u;
        float y  = 0.0f;
#pragma unroll
        for (int s = 0; s < DSTATE; ++s) {
            float e = __expf(dt * a[s]);
            h[s] = e * h[s] + du * sB[s];
            y += h[s] * sC[s];
        }
        y += u * Dd;                                 // skip: D*u
        float g = z * (1.0f / (1.0f + __expf(-z)));  // SiLU gate
        y16[row * DINNER + d] = (_Float16)(y * g);
        __syncthreads();
    }
}

// ---------------- launcher ----------------
extern "C" void kernel_launch(void* const* d_in, const int* in_sizes, int n_in,
                              void* d_out, int out_size, void* d_ws, size_t ws_size,
                              hipStream_t stream) {
    const float* x         = (const float*)d_in[0];   // (8,1024,256)
    const float* in_proj_w = (const float*)d_in[1];   // (1024,256)
    const float* conv_w    = (const float*)d_in[2];   // (512,1,4)
    const float* conv_b    = (const float*)d_in[3];   // (512)
    const float* x_proj_w  = (const float*)d_in[4];   // (144,512)
    const float* dt_proj_w = (const float*)d_in[5];   // (512,16)
    const float* dt_proj_b = (const float*)d_in[6];   // (512)
    const float* A_log     = (const float*)d_in[7];   // (512,64)
    const float* Dp        = (const float*)d_in[8];   // (512)
    const float* out_proj_w= (const float*)d_in[9];   // (256,512)
    const float* lin_w     = (const float*)d_in[10];  // (256,256)
    const float* lin_b     = (const float*)d_in[11];  // (256)
    float* out = (float*)d_out;                       // (8,1024,256)

    // ---- workspace carve-out (all 256B aligned) ----
    char* ws = (char*)d_ws;
    size_t off = 0;
    auto carve = [&](size_t bytes) -> char* {
        char* p = ws + off;
        off = (off + bytes + 255) & ~(size_t)255;
        return p;
    };
    _Float16* X16    = (_Float16*)carve((size_t)MROWS * DMODEL * 2);
    _Float16* WIN16  = (_Float16*)carve((size_t)NXZ * DMODEL * 2);
    _Float16* XPW16  = (_Float16*)carve((size_t)NXD * DINNER * 2);
    _Float16* DTW16  = (_Float16*)carve((size_t)DINNER * 32 * 2);
    _Float16* OPW16  = (_Float16*)carve((size_t)DMODEL * DINNER * 2);
    _Float16* LINW16 = (_Float16*)carve((size_t)DMODEL * DMODEL * 2);
    float*    XZ     = (float*)   carve((size_t)MROWS * NXZ * 4);
    float*    UCONV  = (float*)   carve((size_t)MROWS * DINNER * 4);
    _Float16* UCONV16= (_Float16*)carve((size_t)MROWS * DINNER * 2);
    float*    XDBL   = (float*)   carve((size_t)MROWS * NXD * 4);
    _Float16* DT16   = (_Float16*)carve((size_t)MROWS * 32 * 2);
    float*    DELTA  = (float*)   carve((size_t)MROWS * DINNER * 4);
    _Float16* Y16    = (_Float16*)carve((size_t)MROWS * DINNER * 2);
    _Float16* MOUT16 = (_Float16*)carve((size_t)MROWS * DMODEL * 2);

    auto cvt = [&](const float* src, _Float16* dst, int n) {
        k_f32_to_f16<<<(n + 255) / 256, 256, 0, stream>>>(src, dst, n);
    };
    auto gemm4 = [&](const _Float16* A, const _Float16* W, float* Cf, _Float16* Ch,
                     const float* bias, int M, int N, int K, int act) {
        int tiles = (M >> 4) * ((N >> 4) / 4);
        k_wmma_gemm_bt<4><<<(tiles + 7) / 8, 256, 0, stream>>>(A, W, Cf, Ch, bias, M, N, K, act);
    };
    auto gemm3 = [&](const _Float16* A, const _Float16* W, float* Cf, _Float16* Ch,
                     const float* bias, int M, int N, int K, int act) {
        int tiles = (M >> 4) * ((N >> 4) / 3);
        k_wmma_gemm_bt<3><<<(tiles + 7) / 8, 256, 0, stream>>>(A, W, Cf, Ch, bias, M, N, K, act);
    };

    // ---- precision conversion of weights & input ----
    cvt(x,          X16,    MROWS * DMODEL);
    cvt(in_proj_w,  WIN16,  NXZ * DMODEL);
    cvt(x_proj_w,   XPW16,  NXD * DINNER);
    cvt(out_proj_w, OPW16,  DMODEL * DINNER);
    cvt(lin_w,      LINW16, DMODEL * DMODEL);
    k_pad_dtw<<<(DINNER * 32 + 255) / 256, 256, 0, stream>>>(dt_proj_w, DTW16);

    // ---- 1: in_proj  xz = x @ in_proj_w.T  (8192x1024x256) ----
    gemm4(X16, WIN16, XZ, nullptr, nullptr, MROWS, NXZ, DMODEL, 0);

    // ---- 2: causal depthwise conv + SiLU on u = xz[:, :512] ----
    k_conv_silu<<<(MROWS * DINNER + 255) / 256, 256, 0, stream>>>(XZ, conv_w, conv_b, UCONV, UCONV16);

    // ---- 3: x_proj  x_dbl = u @ x_proj_w.T  (8192x144x512); 9 N-tiles = 3 strips of 3 ----
    gemm3(UCONV16, XPW16, XDBL, nullptr, nullptr, MROWS, NXD, DINNER, 0);

    // ---- 4: dt_proj + softplus  delta = softplus(dt @ dt_proj_w.T + b) ----
    k_extract_dt<<<(MROWS * 32 + 255) / 256, 256, 0, stream>>>(XDBL, DT16);
    gemm4(DT16, DTW16, DELTA, nullptr, dt_proj_b, MROWS, DINNER, 32, 1);

    // ---- 5: selective scan + D*u skip + SiLU(z) gate ----
    k_ssm_scan<<<BB * (DINNER / SCAN_TPB), SCAN_TPB, 0, stream>>>(
        DELTA, UCONV, XDBL, XZ, A_log, Dp, Y16);

    // ---- 6: out_proj  (8192x256x512), f16 out ----
    gemm4(Y16, OPW16, nullptr, MOUT16, nullptr, MROWS, DMODEL, DINNER, 0);

    // ---- 7: final linear + bias -> d_out (f32) ----
    gemm4(MOUT16, LINW16, out, nullptr, lin_b, MROWS, DMODEL, DMODEL, 0);

    (void)in_sizes; (void)n_in; (void)out_size; (void)ws_size;
}
